// CVQ_15341623181535
// MI455X (gfx1250) — compile-verified, matrix operand order checked by reading, not run on hardware
//
#include <hip/hip_runtime.h>
#include <hip/hip_bf16.h>

// CVQ fused forward for gfx1250 (MI455X).
// - v_wmma_f32_16x16x32_f16 for the zn @ cn^T GEMM (K=64 -> 2 WMMA/tile)
// - branchless packed-u64 argmax (no exec-mask branches in the hot loop)
// - codebook staged via LDS with global_load_async_to_lds_b128 + s_wait_asynccnt
// - ds_add_f32 / ds_max_u64 LDS atomics for per-column stats

typedef __attribute__((ext_vector_type(16))) _Float16 v16h;
typedef __attribute__((ext_vector_type(8)))  float    v8f;
typedef __attribute__((ext_vector_type(4)))  int      v4i;

#define NUM_EMBED 1024
#define EMBED_DIM 64
#define B_ROWS    65536
#define BETA      0.25f
#define DECAY     0.99f
#define EPSN      1e-12f
#define INV_TEMP  (1.0f/0.07f)

#if defined(__has_builtin)
#if __has_builtin(__builtin_amdgcn_global_load_async_to_lds_b128)
#define HAVE_ASYNC_LDS 1
#endif
#endif

// monotonic f32 -> u32 key (branchless): preserves float ordering in unsigned
__device__ __forceinline__ unsigned ordered_u32(float f) {
  unsigned u = __float_as_uint(f);
  return u ^ ((unsigned)((int)u >> 31) | 0x80000000u);
}
__device__ __forceinline__ unsigned long long umax64(unsigned long long a,
                                                     unsigned long long b) {
  return a > b ? a : b;   // v_cmp_lt_u64 + cndmask pair, branchless
}
__device__ __forceinline__ unsigned long long shfl_xor_u64(unsigned long long v,
                                                           int mask) {
  unsigned lo = (unsigned)v, hi = (unsigned)(v >> 32);
  lo = __shfl_xor(lo, mask);
  hi = __shfl_xor(hi, mask);
  return ((unsigned long long)hi << 32) | lo;
}

// ---------------- Kernel 0: normalize codebook rows -> f16, zero accumulators
__global__ void cvq_k0_init(const float* __restrict__ w,
                            _Float16* __restrict__ cn,
                            unsigned long long* __restrict__ g_pack,
                            float* __restrict__ g_sum,
                            float* __restrict__ counts,
                            float* __restrict__ scalars /*[0]=sq,[1]=contra*/) {
  __shared__ float red[64];
  int n = blockIdx.x, k = threadIdx.x;
  float v = w[n * EMBED_DIM + k];
  red[k] = v * v;
  __syncthreads();
  for (int s = 32; s > 0; s >>= 1) {
    if (k < s) red[k] += red[k + s];
    __syncthreads();
  }
  float scale = 1.0f / fmaxf(sqrtf(red[0]), EPSN);
  cn[n * EMBED_DIM + k] = (_Float16)(v * scale);
  if (k == 0) { g_pack[n] = 0ull; g_sum[n] = 0.0f; counts[n] = 0.0f; }
  if (n == 0 && k < 2) scalars[k] = 0.0f;
}

// ---------------- Kernel 1: fused WMMA GEMM + row/column statistics
__global__ __launch_bounds__(256) void cvq_k1_gemm(
    const float* __restrict__ z,
    const _Float16* __restrict__ cn,
    float* __restrict__ out_idx,
    unsigned long long* __restrict__ g_pack,
    float* __restrict__ g_sum) {
  __shared__ _Float16 sh_cn[256 * EMBED_DIM];          // 32 KB
  __shared__ unsigned long long sh_pack[NUM_EMBED];    // 8 KB
  __shared__ float sh_sum[NUM_EMBED];                  // 4 KB

  const int tid  = threadIdx.x;
  const int lane = tid & 31;
  const int r    = lane & 15;
  const int koff = (lane < 16) ? 0 : 8;
  const int mbase = (lane < 16) ? 0 : 8;
  const int rowbase = blockIdx.x * 128 + (tid >> 5) * 16;

  for (int i = tid; i < NUM_EMBED; i += 256) { sh_pack[i] = 0ull; sh_sum[i] = 0.0f; }

  // --- load + normalize this lane's half-row of z, build A fragments
  const float* zp = z + (size_t)(rowbase + r) * EMBED_DIM;
  float buf[32];
  float ss = 0.0f;
#pragma unroll
  for (int j = 0; j < 8; ++j) {
    buf[j]      = zp[koff + j];
    buf[8 + j]  = zp[koff + 16 + j];
    buf[16 + j] = zp[koff + 32 + j];
    buf[24 + j] = zp[koff + 48 + j];
  }
#pragma unroll
  for (int j = 0; j < 32; ++j) ss += buf[j] * buf[j];
  ss += __shfl_xor(ss, 16);                 // partner lane holds other half-row
  float scale = 1.0f / fmaxf(sqrtf(ss), EPSN);
  v16h a0, a1;
#pragma unroll
  for (int j = 0; j < 8; ++j) {
    a0[j]     = (_Float16)(buf[j] * scale);
    a0[8 + j] = (_Float16)(buf[8 + j] * scale);
    a1[j]     = (_Float16)(buf[16 + j] * scale);
    a1[8 + j] = (_Float16)(buf[24 + j] * scale);
  }

  // packed (value, ~index) keys: max key == max value, tie -> smallest index
  unsigned long long rowPk[8];
#pragma unroll
  for (int s = 0; s < 8; ++s) rowPk[s] = 0ull;

  const uint4* cn4 = (const uint4*)cn;
  uint4* sh4 = (uint4*)sh_cn;

  for (int chunk = 0; chunk < 4; ++chunk) {
    __syncthreads();
#if HAVE_ASYNC_LDS
    for (int q = tid; q < 2048; q += 256) {
      __builtin_amdgcn_global_load_async_to_lds_b128(
          (v4i*)(cn4 + chunk * 2048 + q), (v4i*)(sh4 + q), 0, 0);
    }
#if __has_builtin(__builtin_amdgcn_s_wait_asynccnt)
    __builtin_amdgcn_s_wait_asynccnt(0);
#else
    asm volatile("s_wait_asynccnt 0" ::: "memory");
#endif
#else
    for (int q = tid; q < 2048; q += 256) sh4[q] = cn4[chunk * 2048 + q];
#endif
    if (chunk < 3)  // prefetch next codebook chunk toward L2
      __builtin_prefetch(cn4 + (chunk + 1) * 2048 + tid * 8, 0, 1);
    __syncthreads();

    for (int t = 0; t < 16; ++t) {
      const int colbase = chunk * 256 + t * 16;
      const _Float16* bp = &sh_cn[(t * 16 + r) * EMBED_DIM + koff];
      v16h b0, b1;
#pragma unroll
      for (int j = 0; j < 8; ++j) {
        b0[j]     = bp[j];
        b0[8 + j] = bp[16 + j];
        b1[j]     = bp[32 + j];
        b1[8 + j] = bp[48 + j];
      }
      v8f c = {};
      c = __builtin_amdgcn_wmma_f32_16x16x32_f16(false, a0, false, b0, (short)0, c, false, false);
      c = __builtin_amdgcn_wmma_f32_16x16x32_f16(false, a1, false, b1, (short)0, c, false, false);

      // C/D layout: lane -> N (lane&15), VGPR slot s -> M (mbase+s). Branchless stats.
      const int mycol = colbase + r;
      const unsigned notcol = ~(unsigned)mycol;
      unsigned long long colPk = 0ull;
      float csum = 0.0f;
#pragma unroll
      for (int s = 0; s < 8; ++s) {
        float v = c[s];
        unsigned long long hi = ((unsigned long long)ordered_u32(v)) << 32;
        rowPk[s] = umax64(rowPk[s], hi | notcol);
        colPk    = umax64(colPk, hi | (unsigned)~(rowbase + mbase + s));
        csum += __expf((v - 1.0f) * INV_TEMP);     // offset logsumexp accumulator
      }
      // combine with partner lane (same column, other 8 rows)
      csum += __shfl_xor(csum, 16);
      colPk = umax64(colPk, shfl_xor_u64(colPk, 16));
      if (lane < 16) {
        atomicAdd(&sh_sum[mycol], csum);
        atomicMax(&sh_pack[mycol], colPk);
      }
    }
  }

  // --- row argmax butterfly across the 16-lane group (branchless)
#pragma unroll
  for (int m = 1; m < 16; m <<= 1) {
#pragma unroll
    for (int s = 0; s < 8; ++s)
      rowPk[s] = umax64(rowPk[s], shfl_xor_u64(rowPk[s], m));
  }
  if (r == 0) {
    int rb = rowbase + mbase;
#pragma unroll
    for (int s = 0; s < 8; ++s)
      out_idx[rb + s] = (float)(~(unsigned)rowPk[s]);
  }

  // --- flush column accumulators to global
  __syncthreads();
  for (int i = tid; i < NUM_EMBED; i += 256) {
    atomicAdd(&g_sum[i], sh_sum[i]);
    atomicMax(&g_pack[i], sh_pack[i]);
  }
}

// ---------------- Kernel 2: z_q_st, counts, MSE accumulation
__global__ __launch_bounds__(256) void cvq_k2_rows(
    const float* __restrict__ z,
    const float* __restrict__ w,
    const float* __restrict__ idxf,
    float* __restrict__ zq_out,
    float* __restrict__ counts,
    float* __restrict__ scalars) {
  __shared__ float red[256];
  int gid = blockIdx.x * 256 + threadIdx.x;
  int row = gid >> 6;
  int k   = gid & 63;
  int code = (int)idxf[row];
  float zq = w[code * EMBED_DIM + k];
  float zv = z[gid];
  zq_out[gid] = zv + (zq - zv);        // straight-through forward value
  float d = zq - zv;
  red[threadIdx.x] = d * d;
  if (k == 0) atomicAdd(&counts[code], 1.0f);
  __syncthreads();
  for (int s = 128; s > 0; s >>= 1) {
    if (threadIdx.x < s) red[threadIdx.x] += red[threadIdx.x + s];
    __syncthreads();
  }
  if (threadIdx.x == 0) atomicAdd(&scalars[0], red[0]);
}

// ---------------- Kernel 3: EMA probs, anchored weight update, contra term
__global__ void cvq_k3_codes(
    const float* __restrict__ z,
    const float* __restrict__ w,
    const float* __restrict__ prob,
    const float* __restrict__ counts,
    const unsigned long long* __restrict__ g_pack,
    const float* __restrict__ g_sum,
    float* __restrict__ out_neww,
    float* __restrict__ out_prob,
    float* __restrict__ scalars) {
  __shared__ float s_dec;
  __shared__ int s_row;
  int n = blockIdx.x, k = threadIdx.x;
  if (k == 0) {
    float avg = counts[n] / (float)B_ROWS;
    float pn  = prob[n] * DECAY + avg * (1.0f - DECAY);
    out_prob[n] = pn;
    float dec = __expf(-(pn * (float)NUM_EMBED * 10.0f) / (1.0f - DECAY) - 0.001f);
    unsigned long long pk = g_pack[n];
    int row = (int)(~(unsigned)(pk & 0xFFFFFFFFull));
    unsigned ou = (unsigned)(pk >> 32);
    unsigned u  = (ou & 0x80000000u) ? (ou & 0x7FFFFFFFu) : ~ou;
    float maxd  = __uint_as_float(u);
    // logsumexp(d/T) = log(sum exp((d-1)/T)) + 1/T ; dis_pos ~ column max
    float contra = logf(fmaxf(g_sum[n], 1e-30f)) + (1.0f - maxd) * INV_TEMP;
    atomicAdd(&scalars[1], contra);
    s_dec = dec; s_row = row;
  }
  __syncthreads();
  float dec = s_dec;
  int row = s_row;
  float wv = w[n * EMBED_DIM + k];
  float rf = z[(size_t)row * EMBED_DIM + k];
  out_neww[n * EMBED_DIM + k] = wv * (1.0f - dec) + rf * dec;
}

// ---------------- Kernel 4: final scalar loss
__global__ void cvq_k4_loss(const float* __restrict__ scalars,
                            float* __restrict__ out_loss) {
  out_loss[0] = (1.0f + BETA) * scalars[0] / (float)(B_ROWS * EMBED_DIM) +
                scalars[1] / (float)NUM_EMBED;
}

extern "C" void kernel_launch(void* const* d_in, const int* in_sizes, int n_in,
                              void* d_out, int out_size, void* d_ws, size_t ws_size,
                              hipStream_t stream) {
  const float* z    = (const float*)d_in[0];
  const float* w    = (const float*)d_in[1];
  const float* prob = (const float*)d_in[2];
  float* out = (float*)d_out;

  const size_t Z_N = (size_t)B_ROWS * EMBED_DIM;   // 4,194,304
  float* out_z    = out;
  float* out_loss = out + Z_N;
  float* out_idx  = out + Z_N + 1;
  float* out_neww = out + Z_N + 1 + B_ROWS;
  float* out_prob = out + Z_N + 1 + B_ROWS + (size_t)NUM_EMBED * EMBED_DIM;

  char* ws = (char*)d_ws;
  _Float16* cn                 = (_Float16*)ws;                          // 128 KB
  unsigned long long* g_pack   = (unsigned long long*)(ws + 131072);     // 8 KB
  float* g_sum                 = (float*)(ws + 139264);                  // 4 KB
  float* counts                = (float*)(ws + 143360);                  // 4 KB
  float* scalars               = (float*)(ws + 147456);                  // 8 B

  hipLaunchKernelGGL(cvq_k0_init,  dim3(NUM_EMBED), dim3(64), 0, stream,
                     w, cn, g_pack, g_sum, counts, scalars);
  hipLaunchKernelGGL(cvq_k1_gemm,  dim3(B_ROWS / 128), dim3(256), 0, stream,
                     z, cn, out_idx, g_pack, g_sum);
  hipLaunchKernelGGL(cvq_k2_rows,  dim3((unsigned)(Z_N / 256)), dim3(256), 0, stream,
                     z, w, out_idx, out_z, counts, scalars);
  hipLaunchKernelGGL(cvq_k3_codes, dim3(NUM_EMBED), dim3(64), 0, stream,
                     z, w, prob, counts, g_pack, g_sum, out_neww, out_prob, scalars);
  hipLaunchKernelGGL(cvq_k4_loss,  dim3(1), dim3(1), 0, stream, scalars, out_loss);
}
